// LSTM_37632503447943
// MI455X (gfx1250) — matrix operand order repaired and checked by reading, not compiled, not run on hardware
//
#include <hip/hip_runtime.h>
#include <hip/hip_bf16.h>
#include <math.h>

// Problem constants (match reference)
#define VV 32000
#define EE 1024
#define HH 1024
#define BB 16
#define TT 256
#define G4H 4096   // 4*H
#define SCAN_BLOCKS 8u   // 8 blocks x 256 thr = 64 waves = H/16 hidden tiles

typedef __attribute__((ext_vector_type(16))) _Float16 v16h;
typedef __attribute__((ext_vector_type(8)))  _Float16 v8h;
typedef __attribute__((ext_vector_type(8)))  float    v8f;

// ---------------------------------------------------------------------------
// Elementwise helpers
// ---------------------------------------------------------------------------
__global__ void cast_f32_f16(const float* __restrict__ s, _Float16* __restrict__ d, int n) {
    int i = blockIdx.x * blockDim.x + threadIdx.x;
    if (i < n) d[i] = (_Float16)s[i];
}

__global__ void zero_bytes(unsigned int* __restrict__ p, int nwords) {
    int i = blockIdx.x * blockDim.x + threadIdx.x;
    if (i < nwords) p[i] = 0u;
}

// Gather embedding rows: X[t][b][e] = emb[x_ids[b][t]][e], cast to f16
__global__ void embed_gather_f16(const int* __restrict__ ids, const float* __restrict__ emb,
                                 _Float16* __restrict__ X, int total) {
    int idx = blockIdx.x * blockDim.x + threadIdx.x;
    if (idx >= total) return;
    int e  = idx & (EE - 1);
    int tb = idx >> 10;          // / EE
    int b  = tb & (BB - 1);
    int t  = tb >> 4;            // / BB
    int id = ids[b * TT + t];
    X[idx] = (_Float16)emb[(size_t)id * EE + e];
}

// ---------------------------------------------------------------------------
// Generic WMMA GEMM:  C[M,N](f32) = A[M,K](f16,row-major) @ Bt[N,K]^T (f16)
// + optional per-column biases.  One wave computes a 16x64 output strip
// (4 N-tiles) reusing the A fragment 4x.  fc_remap!=0 writes to the
// [B,T,V] layout expected by d_out (rows are r = t*B + b).
// ---------------------------------------------------------------------------
__global__ __launch_bounds__(256) void gemm_tn_f16(
    const _Float16* __restrict__ A, const _Float16* __restrict__ Bt,
    const float* __restrict__ bias1, const float* __restrict__ bias2,
    float* __restrict__ C, int M, int N, int K, int fc_remap)
{
    int wid  = (int)((blockIdx.x * blockDim.x + threadIdx.x) >> 5);
    int lane = threadIdx.x & 31;
    int ngroups = N >> 6;                 // N/64
    int total   = (M >> 4) * ngroups;
    if (wid >= total) return;
    int mtile = wid / ngroups;
    int ng    = wid % ngroups;
    int lh      = lane & 15;
    int hiHalf  = lane >> 4;              // 0: lanes 0-15, 1: lanes 16-31

    v8f acc[4];
    #pragma unroll
    for (int j = 0; j < 4; ++j) {
        int col = ng * 64 + j * 16 + lh;
        float bv = 0.f;
        if (bias1) bv += bias1[col];
        if (bias2) bv += bias2[col];
        #pragma unroll
        for (int i = 0; i < 8; ++i) acc[j][i] = bv;
    }

    // A fragment per lane: row M = mtile*16+lh; K = kb+{0..15} (lanes<16)
    //                                           K = kb+{16..31} (lanes>=16)
    const _Float16* arow  = A  + (size_t)(mtile * 16 + lh) * K + hiHalf * 16;
    // B fragment per lane: column N = tile_base+lh; K = kb+{0..15} or kb+{16..31}
    const _Float16* brow0 = Bt + (size_t)(ng * 64 + lh) * K + hiHalf * 16;

    for (int kb = 0; kb < K; kb += 32) {
        __builtin_prefetch(arow + kb + 128, 0, 1);   // global_prefetch_b8
        v16h a;
        *(v8h*)&a       = *(const v8h*)(arow + kb);
        *((v8h*)&a + 1) = *(const v8h*)(arow + kb + 8);
        #pragma unroll
        for (int j = 0; j < 4; ++j) {
            v16h b = *(const v16h*)(brow0 + (size_t)j * 16 * K + kb);
            acc[j] = __builtin_amdgcn_wmma_f32_16x16x32_f16(
                false, a, false, b, (short)0, acc[j], false, false);
        }
    }

    // C/D fragment: VGPR i, lane: element (M = i + 8*hiHalf, N = lh)
    #pragma unroll
    for (int j = 0; j < 4; ++j) {
        int col = ng * 64 + j * 16 + lh;
        #pragma unroll
        for (int i = 0; i < 8; ++i) {
            int row = mtile * 16 + i + hiHalf * 8;
            if (fc_remap) {
                int b = row & (BB - 1);   // rows are r = t*B + b
                int t = row >> 4;
                C[(size_t)b * TT * VV + (size_t)t * VV + col] = acc[j][i];
            } else {
                C[(size_t)row * N + col] = acc[j][i];
            }
        }
    }
}

// ---------------------------------------------------------------------------
// Device-wide sense-reversal barrier for the persistent scan kernel
// (8 co-resident workgroups).  Cluster-barrier is issued as a fast-path hint:
// architecturally NOP when ClusterID==0 (non-cluster dispatch).
// ---------------------------------------------------------------------------
__device__ inline void grid_barrier(unsigned* cnt, unsigned* gen, unsigned nb) {
    __threadfence();                 // make this thread's stores device-visible
    __syncthreads();                 // whole block arrived (all stores fenced)
    __builtin_amdgcn_s_cluster_barrier();   // s_barrier_signal/wait -3 (NOP if unclustered)
    if (threadIdx.x == 0) {
        unsigned g = __hip_atomic_load(gen, __ATOMIC_RELAXED, __HIP_MEMORY_SCOPE_AGENT);
        unsigned arrived = __hip_atomic_fetch_add(cnt, 1u, __ATOMIC_ACQ_REL,
                                                  __HIP_MEMORY_SCOPE_AGENT);
        if (arrived == nb - 1u) {
            __hip_atomic_store(cnt, 0u, __ATOMIC_RELAXED, __HIP_MEMORY_SCOPE_AGENT);
            __hip_atomic_fetch_add(gen, 1u, __ATOMIC_RELEASE, __HIP_MEMORY_SCOPE_AGENT);
        } else {
            while (__hip_atomic_load(gen, __ATOMIC_ACQUIRE,
                                     __HIP_MEMORY_SCOPE_AGENT) == g)
                __builtin_amdgcn_s_sleep(2);
        }
    }
    __syncthreads();
    __threadfence();                 // discard stale lines before next-step reads
}

// ---------------------------------------------------------------------------
// Persistent LSTM layer scan: ONE launch runs all T=256 steps.
// 64 waves; wave w owns hidden units [16w,16w+16) of all four gates, so the
// cell nonlinearity is wave-local.  Cell state lives in registers for the
// entire sequence; only h (needed as next step's A matrix by every wave)
// round-trips through memory, with a device barrier per step.
// ---------------------------------------------------------------------------
__global__ __launch_bounds__(256) void lstm_layer_scan(
    const float*    __restrict__ Gx,     // [T, 16, 4H] hoisted input gates (+biases)
    const _Float16* __restrict__ Whh,    // [4H, H] f16 (row-major = B^T)
    const _Float16* __restrict__ h0,     // [16, H] zeros
    _Float16*       __restrict__ hseq,   // [T, 16, H] f16 out
    unsigned* cnt, unsigned* gen)
{
    int wid  = (int)((blockIdx.x * blockDim.x + threadIdx.x) >> 5);  // 0..63
    int lane = threadIdx.x & 31;
    int lh     = lane & 15;
    int hiHalf = lane >> 4;
    int nh   = wid * 16;
    int hidx = nh + lh;

    float creg[8];                       // persistent cell state (this lane's 8 rows)
    #pragma unroll
    for (int i = 0; i < 8; ++i) creg[i] = 0.f;

    // Per-lane W_hh fragment base pointers (fixed across steps)
    const _Float16* bbase[4];
    #pragma unroll
    for (int g = 0; g < 4; ++g)
        bbase[g] = Whh + (size_t)(g * HH + nh + lh) * HH + hiHalf * 16;

    const _Float16* hprev = h0;
    for (int t = 0; t < TT; ++t) {
        const float* gx = Gx + (size_t)t * BB * G4H;

        v8f acc[4];
        #pragma unroll
        for (int g = 0; g < 4; ++g) {
            int col = g * HH + nh + lh;
            #pragma unroll
            for (int i = 0; i < 8; ++i)
                acc[g][i] = gx[(size_t)(i + hiHalf * 8) * G4H + col];
        }

        const _Float16* arow = hprev + (size_t)lh * HH + hiHalf * 16;
        for (int kb = 0; kb < HH; kb += 32) {
            v16h a;
            *(v8h*)&a       = *(const v8h*)(arow + kb);
            *((v8h*)&a + 1) = *(const v8h*)(arow + kb + 8);
            #pragma unroll
            for (int g = 0; g < 4; ++g) {
                v16h b = *(const v16h*)(bbase[g] + kb);
                acc[g] = __builtin_amdgcn_wmma_f32_16x16x32_f16(
                    false, a, false, b, (short)0, acc[g], false, false);
            }
        }

        _Float16* hout = hseq + (size_t)t * BB * HH;
        #pragma unroll
        for (int i = 0; i < 8; ++i) {
            int row = i + hiHalf * 8;                    // batch index
            float iv = 1.f / (1.f + __expf(-acc[0][i])); // torch gate order i,f,g,o
            float fv = 1.f / (1.f + __expf(-acc[1][i]));
            float gv = tanhf(acc[2][i]);
            float ov = 1.f / (1.f + __expf(-acc[3][i]));
            float cn = fv * creg[i] + iv * gv;
            creg[i] = cn;
            hout[row * HH + hidx] = (_Float16)(ov * tanhf(cn));
        }

        grid_barrier(cnt, gen, SCAN_BLOCKS);             // h(t) visible everywhere
        hprev = hout;
    }
}

// ---------------------------------------------------------------------------
// In-place row softmax over V=32000; one workgroup per (b,t) row of d_out.
// ---------------------------------------------------------------------------
__global__ __launch_bounds__(256) void softmax_rows(float* __restrict__ out) {
    __shared__ float red[256];
    float* p = out + (size_t)blockIdx.x * VV;
    int tid = threadIdx.x;

    float m = -INFINITY;
    for (int v = tid; v < VV; v += 256) m = fmaxf(m, p[v]);
    red[tid] = m; __syncthreads();
    for (int s = 128; s > 0; s >>= 1) {
        if (tid < s) red[tid] = fmaxf(red[tid], red[tid + s]);
        __syncthreads();
    }
    m = red[0]; __syncthreads();

    float sum = 0.f;
    for (int v = tid; v < VV; v += 256) sum += __expf(p[v] - m);
    red[tid] = sum; __syncthreads();
    for (int s = 128; s > 0; s >>= 1) {
        if (tid < s) red[tid] += red[tid + s];
        __syncthreads();
    }
    float inv = 1.f / red[0];

    for (int v = tid; v < VV; v += 256) p[v] = __expf(p[v] - m) * inv;
}

// ---------------------------------------------------------------------------
// Host orchestration (14 launches total)
// ---------------------------------------------------------------------------
extern "C" void kernel_launch(void* const* d_in, const int* in_sizes, int n_in,
                              void* d_out, int out_size, void* d_ws, size_t ws_size,
                              hipStream_t stream) {
    const int*   x_ids = (const int*)  d_in[0];
    const float* emb   = (const float*)d_in[1];
    const float* W_ih1 = (const float*)d_in[2];
    const float* W_hh1 = (const float*)d_in[3];
    const float* b_ih1 = (const float*)d_in[4];
    const float* b_hh1 = (const float*)d_in[5];
    const float* W_ih2 = (const float*)d_in[6];
    const float* W_hh2 = (const float*)d_in[7];
    const float* b_ih2 = (const float*)d_in[8];
    const float* b_hh2 = (const float*)d_in[9];
    const float* fc_W  = (const float*)d_in[10];
    const float* fc_b  = (const float*)d_in[11];
    float* out = (float*)d_out;

    // Workspace carve-up (256B aligned)
    char* w = (char*)d_ws;
    auto alloc = [&](size_t bytes) -> void* {
        void* p = (void*)w; w += (bytes + 255) & ~(size_t)255; return p;
    };
    _Float16* Xh    = (_Float16*)alloc((size_t)TT * BB * EE * 2);   // 8 MB
    _Float16* Wih1h = (_Float16*)alloc((size_t)G4H * EE * 2);       // 8 MB
    _Float16* Whh1h = (_Float16*)alloc((size_t)G4H * HH * 2);       // 8 MB
    _Float16* Wih2h = (_Float16*)alloc((size_t)G4H * HH * 2);       // 8 MB
    _Float16* Whh2h = (_Float16*)alloc((size_t)G4H * HH * 2);       // 8 MB
    _Float16* fcWh  = (_Float16*)alloc((size_t)VV * HH * 2);        // 64 MB
    float*    G1    = (float*)   alloc((size_t)TT * BB * G4H * 4);  // 64 MB
    float*    G2    = (float*)   alloc((size_t)TT * BB * G4H * 4);  // 64 MB
    _Float16* h1h   = (_Float16*)alloc((size_t)TT * BB * HH * 2);   // 8 MB
    _Float16* h2h   = (_Float16*)alloc((size_t)TT * BB * HH * 2);   // 8 MB
    _Float16* h0    = (_Float16*)alloc((size_t)BB * HH * 2);
    unsigned* bar   = (unsigned*)alloc(256);                        // cnt, gen
    (void)ws_size; (void)in_sizes; (void)n_in; (void)out_size;

    const int thr = 256;
    auto blk = [](size_t n, int t) { return (unsigned)((n + t - 1) / t); };

    // 1) casts + gather + state init (self-initializing every call)
    cast_f32_f16<<<blk((size_t)G4H * EE, thr), thr, 0, stream>>>(W_ih1, Wih1h, G4H * EE);
    cast_f32_f16<<<blk((size_t)G4H * HH, thr), thr, 0, stream>>>(W_hh1, Whh1h, G4H * HH);
    cast_f32_f16<<<blk((size_t)G4H * HH, thr), thr, 0, stream>>>(W_ih2, Wih2h, G4H * HH);
    cast_f32_f16<<<blk((size_t)G4H * HH, thr), thr, 0, stream>>>(W_hh2, Whh2h, G4H * HH);
    cast_f32_f16<<<blk((size_t)VV * HH, thr), thr, 0, stream>>>(fc_W, fcWh, VV * HH);
    embed_gather_f16<<<blk((size_t)TT * BB * EE, thr), thr, 0, stream>>>(
        x_ids, emb, Xh, TT * BB * EE);
    zero_bytes<<<blk((size_t)BB * HH / 2 + 64, thr), thr, 0, stream>>>(
        (unsigned*)h0, BB * HH / 2);
    zero_bytes<<<1, 64, 0, stream>>>(bar, 64);   // barrier cnt+gen

    // 2) hoisted input-gate GEMM layer 1: G1 = X @ W_ih1^T + b_ih1 + b_hh1
    {
        unsigned waves = (TT * BB / 16) * (G4H / 64);   // 16384
        gemm_tn_f16<<<waves / 8, thr, 0, stream>>>(
            Xh, Wih1h, b_ih1, b_hh1, G1, TT * BB, G4H, EE, 0);
    }

    // 3) layer-1 recurrence: single persistent launch over all T steps
    lstm_layer_scan<<<SCAN_BLOCKS, thr, 0, stream>>>(
        G1, Whh1h, h0, h1h, bar, bar + 1);

    // 4) hoisted input-gate GEMM layer 2: G2 = h1 @ W_ih2^T + b_ih2 + b_hh2
    {
        unsigned waves = (TT * BB / 16) * (G4H / 64);
        gemm_tn_f16<<<waves / 8, thr, 0, stream>>>(
            h1h, Wih2h, b_ih2, b_hh2, G2, TT * BB, G4H, HH, 0);
    }

    // 5) layer-2 recurrence (barrier cnt self-resets to 0; gen is read-relative)
    lstm_layer_scan<<<SCAN_BLOCKS, thr, 0, stream>>>(
        G2, Whh2h, h0, h2h, bar, bar + 1);

    // 6) FC GEMM straight into d_out with [T,B]->[B,T] remap (+fc_b)
    {
        unsigned waves = (TT * BB / 16) * (VV / 64);    // 128000
        gemm_tn_f16<<<waves / 8, thr, 0, stream>>>(
            h2h, fcWh, fc_b, nullptr, out, TT * BB, VV, HH, 1);
    }

    // 7) in-place softmax over vocabulary
    softmax_rows<<<BB * TT, thr, 0, stream>>>(out);
}
